// FingerPrint_1941325218502
// MI455X (gfx1250) — matrix operand order repaired and checked by reading, not compiled, not run on hardware
//
#include <hip/hip_runtime.h>
#include <hip/hip_bf16.h>

// ---------------------------------------------------------------------------
// Problem constants (from the reference)
// ---------------------------------------------------------------------------
#define BATCH   8192
#define NSLICE  32
#define EDIM    128
#define DDIM    4096      // NSLICE * EDIM
#define HID     1500
#define HIDP    1536      // HID padded to a multiple of 128

typedef __attribute__((ext_vector_type(16))) _Float16 v16h;
typedef __attribute__((ext_vector_type(8)))  float    v8f;

// ---------------------------------------------------------------------------
// Elementwise f32 -> f16 conversion (vectorized x4)
// ---------------------------------------------------------------------------
struct alignas(8) H4 { _Float16 x, y, z, w; };

__global__ void fp1941_cvt_f16_vec4(const float* __restrict__ in,
                                    _Float16* __restrict__ out, int n4) {
    int i = blockIdx.x * blockDim.x + threadIdx.x;
    if (i < n4) {
        float4 f = ((const float4*)in)[i];
        H4 h = { (_Float16)f.x, (_Float16)f.y, (_Float16)f.z, (_Float16)f.w };
        ((H4*)out)[i] = h;
    }
}

// Pad columns: out[r][c] (cOut) = (c < cIn) ? in[r][c] : 0, converted to f16
__global__ void fp1941_pad_cols_f16(const float* __restrict__ in,
                                    _Float16* __restrict__ out,
                                    int rows, int cIn, int cOut) {
    int i = blockIdx.x * blockDim.x + threadIdx.x;
    if (i < rows * cOut) {
        int r = i / cOut, c = i - r * cOut;
        out[i] = (c < cIn) ? (_Float16)in[r * cIn + c] : (_Float16)0.0f;
    }
}

// Pad rows: out[r][c] (rOut rows) = (r < rIn) ? in[r][c] : 0, converted to f16
__global__ void fp1941_pad_rows_f16(const float* __restrict__ in,
                                    _Float16* __restrict__ out,
                                    int rOut, int rIn, int cols) {
    int i = blockIdx.x * blockDim.x + threadIdx.x;
    if (i < rOut * cols) {
        int r = i / cols;
        out[i] = (r < rIn) ? (_Float16)in[i] : (_Float16)0.0f;
    }
}

// Pad a f32 bias vector with zeros
__global__ void fp1941_pad_bias_f32(const float* __restrict__ in,
                                    float* __restrict__ out, int nIn, int nOut) {
    int i = blockIdx.x * blockDim.x + threadIdx.x;
    if (i < nOut) out[i] = (i < nIn) ? in[i] : 0.0f;
}

// ---------------------------------------------------------------------------
// Tiled WMMA GEMM:  C[z] = act( A[z] (MxK, f16) * B[z] (KxN, f16) + bias[z] )
// BM=128, BN=128, BK=64; 256 threads (8 waves); wave tile 32x64 (2x4 WMMA
// per 32-K step, two steps per LDS stage). Double-buffered LDS (64 KB),
// staged with gfx1250 async global->LDS copies (ASYNCcnt), f32 accumulation
// via v_wmma_f32_16x16x32_f16.
// Requires M % 128 == 0, N % 128 == 0, K % 64 == 0 (true for all stages).
// ---------------------------------------------------------------------------
#define BM 128
#define BN 128
#define BK 64

template <bool LEAKY, typename OutT>
__global__ __launch_bounds__(256)
void fp1941_gemm_wmma(const _Float16* __restrict__ A, int lda, long aBatch,
                      const _Float16* __restrict__ Bm, int ldb, long bBatch,
                      const float* __restrict__ bias, long biasBatch,
                      OutT* __restrict__ C, int ldc, long cBatch,
                      int M, int N, int K) {
    __shared__ _Float16 As[2][BM * BK];   // 2 x 16 KB
    __shared__ _Float16 Bs[2][BK * BN];   // 2 x 16 KB

    const int z = blockIdx.z;
    A    += (long)z * aBatch;
    Bm   += (long)z * bBatch;
    bias += (long)z * biasBatch;
    C    += (long)z * cBatch;

    const int rowB = blockIdx.y * BM;
    const int colB = blockIdx.x * BN;

    const int t    = threadIdx.x;
    const int lane = t & 31;
    const int wid  = t >> 5;
    const int waveM = (wid & 3) * 32;   // 4 waves along M, 32 rows each
    const int waveN = (wid >> 2) * 64;  // 2 waves along N, 64 cols each

    // LDS byte offsets (low 32 bits of the generic pointer == LDS offset)
    const unsigned asBase = (unsigned)(uintptr_t)&As[0][0];
    const unsigned bsBase = (unsigned)(uintptr_t)&Bs[0][0];

    // Async staging map: each thread moves 64B of A and 64B of B per stage
    const int arow = t >> 1;           // 0..127  A tile row
    const int acol = (t & 1) * 32;     // 0/32    A col base (halves)
    const int brow = t >> 2;           // 0..63   B tile row
    const int bcol = (t & 3) * 32;     // 0..96   B col base (halves)

    auto stageAsync = [&](int kb, int buf) {
        const _Float16* ag = A + (long)(rowB + arow) * lda + kb + acol;
        unsigned al = asBase + (unsigned)buf * (BM * BK * 2)
                    + (unsigned)(arow * BK + acol) * 2;
        const _Float16* bg = Bm + (long)(kb + brow) * ldb + colB + bcol;
        unsigned bl = bsBase + (unsigned)buf * (BK * BN * 2)
                    + (unsigned)(brow * BN + bcol) * 2;
#pragma unroll
        for (int c = 0; c < 4; ++c) {     // 4 x 16B chunks of A
            const void* g = ag + c * 8;
            unsigned    l = al + c * 16;
            asm volatile("global_load_async_to_lds_b128 %0, %1, off"
                         :: "v"(l), "v"(g) : "memory");
        }
#pragma unroll
        for (int c = 0; c < 4; ++c) {     // 4 x 16B chunks of B
            const void* g = bg + c * 8;
            unsigned    l = bl + c * 16;
            asm volatile("global_load_async_to_lds_b128 %0, %1, off"
                         :: "v"(l), "v"(g) : "memory");
        }
    };

    v8f acc[2][4];
#pragma unroll
    for (int mt = 0; mt < 2; ++mt)
#pragma unroll
        for (int nt = 0; nt < 4; ++nt) acc[mt][nt] = (v8f)0.0f;

    const int sel = lane >> 4;  // K-half selector (A layout, ISA 7.12.2)
    const int mr  = lane & 15;  // M row within 16 (A) / N col within 16 (C/D)

    stageAsync(0, 0);
    for (int kb = 0; kb < K; kb += BK) {
        const int buf = (kb >> 6) & 1;
        // This wave's async copies into `buf` complete, then workgroup sync.
        asm volatile("s_wait_asynccnt 0x0" ::: "memory");
        __syncthreads();
        if (kb + BK < K) stageAsync(kb + BK, buf ^ 1);  // overlaps with WMMA

#pragma unroll
        for (int ks = 0; ks < BK; ks += 32) {
            union { v16h v; uint4 q[2]; } af[2], bf[4];
#pragma unroll
            for (int mt = 0; mt < 2; ++mt) {
                const _Float16* ap = &As[buf][(waveM + mt * 16 + mr) * BK + ks];
                af[mt].q[0] = *(const uint4*)(ap + sel * 8);      // K=sel*8..
                af[mt].q[1] = *(const uint4*)(ap + 16 + sel * 8); // K=16+sel*8..
            }
#pragma unroll
            for (int nt = 0; nt < 4; ++nt) {
                const _Float16* bp = &Bs[buf][ks * BN + waveN + nt * 16 + mr];
#pragma unroll
                for (int s = 0; s < 8; ++s) {
                    bf[nt].v[s]     = bp[(sel * 8 + s) * BN];
                    bf[nt].v[s + 8] = bp[(16 + sel * 8 + s) * BN];
                }
            }
#pragma unroll
            for (int mt = 0; mt < 2; ++mt)
#pragma unroll
                for (int nt = 0; nt < 4; ++nt)
                    acc[mt][nt] = __builtin_amdgcn_wmma_f32_16x16x32_f16(
                        false, af[mt].v, false, bf[nt].v,
                        (short)0, acc[mt][nt], false, false);
        }
    }

    // Epilogue: bias + optional LeakyReLU; C/D layout: N = lane&15, M = sel*8+j
#pragma unroll
    for (int nt = 0; nt < 4; ++nt) {
        const int col = colB + waveN + nt * 16 + mr;
        const float bval = bias[col];
#pragma unroll
        for (int mt = 0; mt < 2; ++mt) {
            const int row0 = rowB + waveM + mt * 16 + sel * 8;
#pragma unroll
            for (int j = 0; j < 8; ++j) {
                float v = acc[mt][nt][j] + bval;
                if (LEAKY) v = (v > 0.0f) ? v : v * 0.01f;
                C[(long)(row0 + j) * ldc + col] = (OutT)v;
            }
        }
    }
}

// ---------------------------------------------------------------------------
// Host-side orchestration
// ---------------------------------------------------------------------------
extern "C" void kernel_launch(void* const* d_in, const int* in_sizes, int n_in,
                              void* d_out, int out_size, void* d_ws, size_t ws_size,
                              hipStream_t stream) {
    // Inputs (setup_inputs order): 0 x, 1 Wq, 2 bq, 3 Wk, 4 bk, 5 Wv, 6 bv,
    //                              7 Wc, 8 bc, 9 W1, 10 b1, 11 W2, 12 b2
    // Softmax over a size-1 axis == 1  =>  Q/K branch is dead; out == v.
    const float* x  = (const float*)d_in[0];
    const float* Wv = (const float*)d_in[5];
    const float* bv = (const float*)d_in[6];
    const float* Wc = (const float*)d_in[7];
    const float* bc = (const float*)d_in[8];
    const float* W1 = (const float*)d_in[9];
    const float* b1 = (const float*)d_in[10];
    const float* W2 = (const float*)d_in[11];
    const float* b2 = (const float*)d_in[12];
    float* y = (float*)d_out;

    // Workspace carving (256B aligned)
    size_t off = 0;
    auto carve = [&](size_t bytes) -> void* {
        void* p = (char*)d_ws + off;
        off = (off + bytes + 255) & ~(size_t)255;
        return p;
    };
    _Float16* xh  = (_Float16*)carve((size_t)BATCH * DDIM * 2);  // x (f16); reused as c
    _Float16* vh  = (_Float16*)carve((size_t)BATCH * DDIM * 2);  // v (f16)
    _Float16* hh  = (_Float16*)carve((size_t)BATCH * HIDP * 2);  // h (f16, padded)
    _Float16* Wvh = (_Float16*)carve((size_t)NSLICE * EDIM * EDIM * 2);
    _Float16* Wch = (_Float16*)carve((size_t)DDIM * DDIM * 2);
    _Float16* W1h = (_Float16*)carve((size_t)DDIM * HIDP * 2);   // col-padded
    _Float16* W2h = (_Float16*)carve((size_t)HIDP * EDIM * 2);   // row-padded
    float*    b1p = (float*)   carve((size_t)HIDP * 4);
    _Float16* ch  = xh;  // x dead after stage 1 -> alias

    const int T = 256;

    // --- precision conversions -------------------------------------------
    {
        int n4 = (BATCH * DDIM) / 4;
        fp1941_cvt_f16_vec4<<<(n4 + T - 1) / T, T, 0, stream>>>(x, xh, n4);
    }
    {
        int n4 = (NSLICE * EDIM * EDIM) / 4;
        fp1941_cvt_f16_vec4<<<(n4 + T - 1) / T, T, 0, stream>>>(Wv, Wvh, n4);
    }
    {
        int n4 = (DDIM * DDIM) / 4;
        fp1941_cvt_f16_vec4<<<(n4 + T - 1) / T, T, 0, stream>>>(Wc, Wch, n4);
    }
    {
        int n = DDIM * HIDP;
        fp1941_pad_cols_f16<<<(n + T - 1) / T, T, 0, stream>>>(W1, W1h, DDIM, HID, HIDP);
    }
    {
        int n = HIDP * EDIM;
        fp1941_pad_rows_f16<<<(n + T - 1) / T, T, 0, stream>>>(W2, W2h, HIDP, HID, EDIM);
    }
    fp1941_pad_bias_f32<<<(HIDP + T - 1) / T, T, 0, stream>>>(b1, b1p, HID, HIDP);

    // --- stage 1: v[:,n,:] = x[:,n,:] @ Wv[n] + bv[n]   (32 batched GEMMs)
    {
        dim3 grid(EDIM / BN, BATCH / BM, NSLICE);
        fp1941_gemm_wmma<false, _Float16><<<grid, T, 0, stream>>>(
            xh, DDIM, (long)EDIM,              // A: column slice per n
            Wvh, EDIM, (long)EDIM * EDIM,      // B: Wv[n]
            bv, (long)EDIM,
            vh, DDIM, (long)EDIM,              // C: column slice per n
            BATCH, EDIM, EDIM);
    }
    // --- stage 2: c = v @ Wc + bc  (8192x4096 x 4096x4096) ----------------
    {
        dim3 grid(DDIM / BN, BATCH / BM, 1);
        fp1941_gemm_wmma<false, _Float16><<<grid, T, 0, stream>>>(
            vh, DDIM, 0, Wch, DDIM, 0, bc, 0,
            ch, DDIM, 0, BATCH, DDIM, DDIM);
    }
    // --- stage 3: h = lrelu(c @ W1 + b1)  (N padded 1500->1536) -----------
    {
        dim3 grid(HIDP / BN, BATCH / BM, 1);
        fp1941_gemm_wmma<true, _Float16><<<grid, T, 0, stream>>>(
            ch, DDIM, 0, W1h, HIDP, 0, b1p, 0,
            hh, HIDP, 0, BATCH, HIDP, DDIM);
    }
    // --- stage 4: y = lrelu(h @ W2 + b2)  (pad rows of W2 are zero) -------
    {
        dim3 grid(EDIM / BN, BATCH / BM, 1);
        fp1941_gemm_wmma<true, float><<<grid, T, 0, stream>>>(
            hh, HIDP, 0, W2h, EDIM, 0, b2, 0,
            y, EDIM, 0, BATCH, EDIM, HIDP);
    }
}
